// HierMatcher_9981503996539
// MI455X (gfx1250) — matrix operand order repaired and checked by reading, not compiled, not run on hardware
//
#include <hip/hip_runtime.h>
#include <hip/hip_bf16.h>
#include <math.h>

// ---------------------------------------------------------------------------
// HierMatcher on gfx1250 (MI455X). Dominant compute: pairwise highway GEMMs
// (131072 x 768) @ (768 x 768) x2 ~ 309 GFLOP -> v_wmma_f32_16x16x32_f16.
// Softmax in token_matching is argmax-only -> elided.
// NOTE: gfx1250 backend crashes selecting fabs on v2f32 (bad i32 bitcast in
// the integer-mask lowering). Any abs idiom (fabsf / fmax(x,-x) / select)
// reconverges there via InstCombine+SLP. Workaround: compute |.| for the f16
// tile purely in the integer domain (trunc->u16, pack, & 0x7fff7fff) with no
// float round-trip, so no llvm.fabs is ever formed in the hot loop.
// ---------------------------------------------------------------------------

#define D_    768
#define B_    4
#define LTOT  128
#define NF    4
#define PITCH 776      // f16 LDS row pitch (16B aligned, bank-conflict skew)
#define NTILE 48       // 768 / 16 N-tiles
#define KSTEP 24       // 768 / 32 K-steps

typedef __attribute__((ext_vector_type(16))) _Float16 v16h;
typedef __attribute__((ext_vector_type(8)))  _Float16 v8h;
typedef __attribute__((ext_vector_type(8)))  float    v8f;

struct FieldPtrs { const float* p[8]; };

union HU { _Float16 h; unsigned short u; };

// ---------------------------------------------------------------------------
// K0: pack 4 fields/side into contiguous (B,128,768) + token masks
// ---------------------------------------------------------------------------
__global__ __launch_bounds__(256) void hm_pack_mask(
    FieldPtrs fp, float* __restrict__ packedL, float* __restrict__ packedR,
    float* __restrict__ tmaskL, float* __restrict__ tmaskR)
{
    const int id = blockIdx.x;            // 2*4*128 = 1024 tokens
    const int side = id >> 9;
    const int b = (id >> 7) & 3;
    const int t = id & 127;
    int f, loc;
    if (t < 32)       { f = 0; loc = t; }
    else if (t < 48)  { f = 1; loc = t - 32; }
    else if (t < 112) { f = 2; loc = t - 48; }
    else              { f = 3; loc = t - 112; }
    constexpr int lens[4] = {32, 16, 64, 16};
    const float* src = fp.p[side * 4 + f] + ((size_t)b * lens[f] + loc) * D_;
    float* dst = (side == 0 ? packedL : packedR) + ((size_t)(b * LTOT + t)) * D_;

    int nz = 0;
    for (int d = threadIdx.x; d < D_; d += 256) {
        float v = src[d];
        dst[d] = v;
        nz |= (v != 0.0f);
    }
    __shared__ int s_nz;
    if (threadIdx.x == 0) s_nz = 0;
    __syncthreads();
    if (nz) atomicOr(&s_nz, 1);
    __syncthreads();
    if (threadIdx.x == 0)
        (side == 0 ? tmaskL : tmaskR)[b * LTOT + t] = s_nz ? 1.0f : 0.0f;
}

// ---------------------------------------------------------------------------
// K1: attribute masks (any token in field present)
// ---------------------------------------------------------------------------
__global__ void hm_amask(const float* __restrict__ tmaskL,
                         const float* __restrict__ tmaskR,
                         float* __restrict__ amaskL, float* __restrict__ amaskR)
{
    constexpr int lens[4]   = {32, 16, 64, 16};
    constexpr int starts[4] = {0, 32, 48, 112};
    const int tid = threadIdx.x;
    if (tid >= 32) return;
    const int side = tid >> 4, b = (tid >> 2) & 3, f = tid & 3;
    const float* tmk = side == 0 ? tmaskL : tmaskR;
    float any = 0.0f;
    for (int t = 0; t < lens[f]; ++t)
        any = fmaxf(any, tmk[b * LTOT + starts[f] + t]);
    (side == 0 ? amaskL : amaskR)[b * NF + f] = any;
}

// ---------------------------------------------------------------------------
// K2: re-block hw_w / gate_w into f16 WMMA B-fragment order.
// Fragment (nt, ks): lane holds 16 halves (v=0..7, 2 each):
//   n = nt*16 + (lane&15);  k = ks*32 + (lane<16?0:16) + 2v + h
// ---------------------------------------------------------------------------
__global__ __launch_bounds__(256) void hm_wblock(
    const float* __restrict__ hw_w, const float* __restrict__ gate_w,
    _Float16* __restrict__ W1blk, _Float16* __restrict__ W2blk)
{
    const size_t per = (size_t)D_ * D_;
    size_t gid = (size_t)blockIdx.x * 256 + threadIdx.x;
    if (gid >= 2 * per) return;
    const int mat = gid >= per;
    const size_t rem0 = mat ? gid - per : gid;
    size_t rem = rem0;
    const int h = rem & 1;  rem >>= 1;
    const int v = rem & 7;  rem >>= 3;
    const int lane = rem & 31; rem >>= 5;
    const int ks = (int)(rem % KSTEP);
    const int nt = (int)(rem / KSTEP);
    const int n = nt * 16 + (lane & 15);
    const int k = ks * 32 + ((lane < 16) ? 0 : 16) + 2 * v + h;
    const float* src = mat ? gate_w : hw_w;
    _Float16* dst = mat ? W2blk : W1blk;
    dst[rem0] = (_Float16)src[(size_t)k * D_ + n];
}

// ---------------------------------------------------------------------------
// K3: fused token matching. One wave per (side, b, i).
// Per 16-j tile: x=|l_i - r_j| in LDS f16, 48 N-tiles x 24 K-steps x 2 WMMA,
// highway epilogue -> scalar scores -> masked argmax -> winning cmp row out.
// ---------------------------------------------------------------------------
__global__ __launch_bounds__(64) void hm_tokmatch(
    const float* __restrict__ packedL, const float* __restrict__ packedR,
    const float* __restrict__ tmaskL, const float* __restrict__ tmaskR,
    const _Float16* __restrict__ W1blk, const _Float16* __restrict__ W2blk,
    const float* __restrict__ hw_b, const float* __restrict__ gate_b,
    const float* __restrict__ tok_w,
    float* __restrict__ cmpL, float* __restrict__ cmpR)
{
    __shared__ alignas(16) _Float16 xs[2][16][PITCH];
    __shared__ alignas(16) float ls[2][D_];

    const int w    = threadIdx.x >> 5;
    const int lane = threadIdx.x & 31;
    const int gw   = blockIdx.x * 2 + w;     // 1024 waves total
    const int side = gw >> 9;
    const int b    = (gw >> 7) & 3;
    const int i    = gw & 127;

    const float* Aseq = side == 0 ? packedL : packedR;
    const float* Bseq = side == 0 ? packedR : packedL;
    const float* tm   = side == 0 ? tmaskR  : tmaskL;   // other side's mask
    float* out        = side == 0 ? cmpL    : cmpR;

    const float* arow = Aseq + ((size_t)(b * LTOT + i)) * D_;
    for (int c = lane; c < D_; c += 32) ls[w][c] = arow[c];
    __syncthreads();

    float best = -INFINITY;
    int bestj = 0;
    const int n     = lane & 15;             // A row m == C col N == lane&15
    const int mbase = (lane >> 4) * 8;       // C rows for this half-wave
    const int kb    = (lane < 16) ? 0 : 16;  // A-fragment K base

    for (int jt = 0; jt < 8; ++jt) {
        // ---- build 16x768 |l_i - r_j| tile in f16 (2 lanes per row) ----
        // abs done as integer sign-clear on packed u16 pairs: no llvm.fabs.
        {
            const int row = lane >> 1, half = lane & 1;
            const float4* b4 =
                (const float4*)(Bseq + ((size_t)(b * LTOT + jt * 16 + row)) * D_);
            const float4* l4 = (const float4*)ls[w];
            for (int c4 = half * 96; c4 < half * 96 + 96; ++c4) {
                float4 bv = b4[c4], lv = l4[c4];
                HU h0, h1, h2, h3;
                h0.h = (_Float16)(lv.x - bv.x);
                h1.h = (_Float16)(lv.y - bv.y);
                h2.h = (_Float16)(lv.z - bv.z);
                h3.h = (_Float16)(lv.w - bv.w);
                unsigned int p0 =
                    ((unsigned int)h0.u | ((unsigned int)h1.u << 16)) & 0x7fff7fffu;
                unsigned int p1 =
                    ((unsigned int)h2.u | ((unsigned int)h3.u << 16)) & 0x7fff7fffu;
                uint2 pk; pk.x = p0; pk.y = p1;
                *(uint2*)&xs[w][row][c4 * 4] = pk;
            }
        }
        __syncthreads();

        float sacc[8] = {0, 0, 0, 0, 0, 0, 0, 0};
        for (int nt = 0; nt < NTILE; ++nt) {
            v8f c1 = {}; v8f c2 = {};
            const _Float16* w1p = W1blk + (((size_t)nt * KSTEP) * 32 + lane) * 16;
            const _Float16* w2p = W2blk + (((size_t)nt * KSTEP) * 32 + lane) * 16;
            for (int ks = 0; ks < KSTEP; ++ks) {
                v8h alo = *(const v8h*)&xs[w][n][ks * 32 + kb];
                v8h ahi = *(const v8h*)&xs[w][n][ks * 32 + kb + 8];
                v16h a;
                #pragma unroll
                for (int t = 0; t < 8; ++t) { a[t] = alo[t]; a[8 + t] = ahi[t]; }
                v16h b1 = *(const v16h*)(w1p + (size_t)ks * 512);
                v16h b2 = *(const v16h*)(w2p + (size_t)ks * 512);
                c1 = __builtin_amdgcn_wmma_f32_16x16x32_f16(
                        false, a, false, b1, (short)0, c1, false, false);
                c2 = __builtin_amdgcn_wmma_f32_16x16x32_f16(
                        false, a, false, b2, (short)0, c2, false, false);
            }
            const int col = nt * 16 + n;
            const float hb = hw_b[col], gb = gate_b[col], tw = tok_w[col];
            #pragma unroll
            for (int r = 0; r < 8; ++r) {
                float xv = (float)xs[w][mbase + r][col];
                float nn = fmaxf(c1[r] + hb, 0.0f);
                float gg = 1.0f / (1.0f + expf(-(c2[r] + gb)));
                float hh = nn * gg + (1.0f - gg) * xv;
                sacc[r] += hh * tw;
            }
        }
        // reduce partial scores over the 16 N-lanes
        #pragma unroll
        for (int m = 1; m <= 8; m <<= 1) {
            #pragma unroll
            for (int r = 0; r < 8; ++r) sacc[r] += __shfl_xor(sacc[r], m, 32);
        }
        // masked argmax (first-max semantics: ascending j, strict >)
        #pragma unroll
        for (int r = 0; r < 8; ++r) {
            const int j = jt * 16 + mbase + r;
            const float key = (tm[b * LTOT + j] > 0.0f) ? sacc[r] : -INFINITY;
            if (key > best) { best = key; bestj = j; }
        }
        __syncthreads();
    }
    // wave-wide argmax; ties -> smaller j
    #pragma unroll
    for (int m = 1; m < 32; m <<= 1) {
        float ob = __shfl_xor(best, m, 32);
        int   oj = __shfl_xor(bestj, m, 32);
        if (ob > best || (ob == best && oj < bestj)) { best = ob; bestj = oj; }
    }
    // winning cmp row (strided per lane: stays scalar, no vector fabs)
    const float* brow = Bseq + ((size_t)(b * LTOT + bestj)) * D_;
    float* orow = out + ((size_t)(b * LTOT + i)) * D_;
    for (int c = lane; c < D_; c += 32) {
        float d = ls[w][c] - brow[c];
        orow[c] = fmaxf(d, -d);
    }
}

// ---------------------------------------------------------------------------
// K4: attribute matching. One block per (side, b, field).
// Matches reference softmax exactly (f32 exp, no max subtraction, +1e-7).
// ---------------------------------------------------------------------------
__global__ __launch_bounds__(256) void hm_attrmatch(
    const float* __restrict__ packedL, const float* __restrict__ packedR,
    const float* __restrict__ tmaskL, const float* __restrict__ tmaskR,
    const float* __restrict__ amaskL, const float* __restrict__ amaskR,
    const float* __restrict__ cmpL, const float* __restrict__ cmpR,
    const float* __restrict__ embL, const float* __restrict__ embR,
    const float* __restrict__ empty_attr, float* __restrict__ rep)
{
    constexpr int lens[4]   = {32, 16, 64, 16};
    constexpr int starts[4] = {0, 32, 48, 112};
    const int id = blockIdx.x;               // 32 blocks
    const int side = id >> 4;
    const int b = (id >> 2) & 3;
    const int f = id & 3;
    const int L = lens[f], st = starts[f];
    const float* pk  = side == 0 ? packedL : packedR;
    const float* tmk = side == 0 ? tmaskL  : tmaskR;
    const float* amk = side == 0 ? amaskL  : amaskR;
    const float* cmp = side == 0 ? cmpL    : cmpR;
    const float* emb = (side == 0 ? embL : embR) + (size_t)f * D_;

    __shared__ float sv[64];
    const int t = threadIdx.x >> 2, sub = threadIdx.x & 3;
    if (t < L) {
        const float* v = pk + ((size_t)(b * LTOT + st + t)) * D_;
        float acc = 0.0f;
        for (int d = sub; d < D_; d += 4) acc += v[d] * emb[d];
        acc += __shfl_xor(acc, 1, 4);
        acc += __shfl_xor(acc, 2, 4);
        if (sub == 0) sv[t] = acc;
    }
    __syncthreads();
    if (threadIdx.x == 0) {
        float sum = 0.0f;
        for (int tt = 0; tt < L; ++tt) {
            float e = expf(sv[tt]) * tmk[b * LTOT + st + tt];
            sv[tt] = e; sum += e;
        }
        float inv = 1.0f / (sum + 1e-7f);
        for (int tt = 0; tt < L; ++tt) sv[tt] *= inv;
    }
    __syncthreads();
    const float am = amk[b * NF + f];
    for (int d = threadIdx.x; d < D_; d += 256) {
        float acc = 0.0f;
        for (int tt = 0; tt < L; ++tt)
            acc += cmp[((size_t)(b * LTOT + st + tt)) * D_ + d] * sv[tt];
        rep[(((size_t)b * 8) + side * 4 + f) * D_ + d] =
            (am > 0.0f) ? acc : empty_attr[d];
    }
}

// ---------------------------------------------------------------------------
// K5/K6: classifier head (tiny; plain VALU)
// ---------------------------------------------------------------------------
__global__ __launch_bounds__(256) void hm_mlp1(
    const float* __restrict__ rep, const float* __restrict__ w1,
    const float* __restrict__ b1, float* __restrict__ hid)
{
    const int idx = blockIdx.x * 256 + threadIdx.x;
    if (idx >= B_ * 1600) return;
    const int b = idx / 1600, oc = idx % 1600;
    const float* r = rep + (size_t)b * 6144;
    float acc = b1[oc];
    for (int k = 0; k < 6144; ++k) acc += r[k] * w1[(size_t)k * 1600 + oc];
    hid[idx] = fmaxf(acc, 0.0f);
}

__global__ __launch_bounds__(256) void hm_mlp2(
    const float* __restrict__ hid, const float* __restrict__ w2,
    const float* __restrict__ b2, float* __restrict__ out)
{
    const int pair = threadIdx.x >> 5, lane = threadIdx.x & 31;
    const int b = pair >> 1, o = pair & 1;
    float acc = 0.0f;
    for (int k = lane; k < 1600; k += 32) acc += hid[b * 1600 + k] * w2[k * 2 + o];
    #pragma unroll
    for (int m = 1; m < 32; m <<= 1) acc += __shfl_xor(acc, m, 32);
    if (lane == 0) out[b * 2 + o] = acc + b2[o];
}

// ---------------------------------------------------------------------------
extern "C" void kernel_launch(void* const* d_in, const int* in_sizes, int n_in,
                              void* d_out, int out_size, void* d_ws, size_t ws_size,
                              hipStream_t stream)
{
    FieldPtrs fp;
    for (int k = 0; k < 8; ++k) fp.p[k] = (const float*)d_in[k];
    const float* hw_w       = (const float*)d_in[8];
    const float* hw_b       = (const float*)d_in[9];
    const float* gate_w     = (const float*)d_in[10];
    const float* gate_b     = (const float*)d_in[11];
    const float* tok_w      = (const float*)d_in[12];
    // d_in[13] = tok_b : uniform shift of scores, argmax-invariant -> unused
    const float* embL       = (const float*)d_in[14];
    const float* embR       = (const float*)d_in[15];
    const float* w1         = (const float*)d_in[16];
    const float* b1         = (const float*)d_in[17];
    const float* w2         = (const float*)d_in[18];
    const float* b2         = (const float*)d_in[19];
    const float* empty_attr = (const float*)d_in[20];

    char* ws = (char*)d_ws;
    size_t off = 0;
    auto take = [&](size_t bytes) {
        void* p = ws + off;
        off += (bytes + 255) & ~(size_t)255;
        return p;
    };
    float*    packedL = (float*)take((size_t)B_ * LTOT * D_ * 4);
    float*    packedR = (float*)take((size_t)B_ * LTOT * D_ * 4);
    float*    tmaskL  = (float*)take((size_t)B_ * LTOT * 4);
    float*    tmaskR  = (float*)take((size_t)B_ * LTOT * 4);
    float*    amaskL  = (float*)take((size_t)B_ * NF * 4);
    float*    amaskR  = (float*)take((size_t)B_ * NF * 4);
    _Float16* W1blk   = (_Float16*)take((size_t)D_ * D_ * 2);
    _Float16* W2blk   = (_Float16*)take((size_t)D_ * D_ * 2);
    float*    cmpL    = (float*)take((size_t)B_ * LTOT * D_ * 4);
    float*    cmpR    = (float*)take((size_t)B_ * LTOT * D_ * 4);
    float*    rep     = (float*)take((size_t)B_ * 8 * D_ * 4);
    float*    hid     = (float*)take((size_t)B_ * 1600 * 4);

    hm_pack_mask<<<2 * B_ * LTOT, 256, 0, stream>>>(fp, packedL, packedR, tmaskL, tmaskR);
    hm_amask<<<1, 32, 0, stream>>>(tmaskL, tmaskR, amaskL, amaskR);
    hm_wblock<<<(int)((2 * (size_t)D_ * D_ + 255) / 256), 256, 0, stream>>>(
        hw_w, gate_w, W1blk, W2blk);
    hm_tokmatch<<<512, 64, 0, stream>>>(packedL, packedR, tmaskL, tmaskR,
                                        W1blk, W2blk, hw_b, gate_b, tok_w,
                                        cmpL, cmpR);
    hm_attrmatch<<<32, 256, 0, stream>>>(packedL, packedR, tmaskL, tmaskR,
                                         amaskL, amaskR, cmpL, cmpR,
                                         embL, embR, empty_attr, rep);
    hm_mlp1<<<(B_ * 1600 + 255) / 256, 256, 0, stream>>>(rep, w1, b1, hid);
    hm_mlp2<<<1, 256, 0, stream>>>(hid, w2, b2, (float*)d_out);

    (void)in_sizes; (void)n_in; (void)out_size; (void)ws_size;
}
